// SSMauBlock_8607114461294
// MI455X (gfx1250) — matrix operand order repaired and checked by reading, not compiled, not run on hardware
//
#include <hip/hip_runtime.h>
#include <math.h>
#include <stdint.h>

// ---------------------------------------------------------------------------
// CDNA5 (gfx1250, wave32) Mamba-like complex-SSM block.
// GEMMs: V_WMMA_F32_16X16X32_F16 with PRE-SPLIT hi/lo f16 operands
// (3 WMMAs per 16x16x32 step, ~2^-22 rel err, f32 accumulate).
// The fused theta-GEMM+scan kernel stages A-chunks into LDS with the
// Tensor Data Mover (tensor_load_to_lds + s_wait_tensorcnt).
// ---------------------------------------------------------------------------

typedef __attribute__((ext_vector_type(16))) _Float16     v16h;
typedef __attribute__((ext_vector_type(8)))  _Float16     v8h;
typedef __attribute__((ext_vector_type(8)))  float        v8f;
typedef __attribute__((ext_vector_type(4)))  unsigned int u32x4;
typedef __attribute__((ext_vector_type(8)))  int          i32x8;
typedef __attribute__((ext_vector_type(4)))  int          i32x4;

#define B_SZ   2
#define L_LEN  1024
#define DM     512
#define NSTATE 16

#if defined(__has_builtin)
#if __has_builtin(__builtin_amdgcn_tensor_load_to_lds) && \
    __has_builtin(__builtin_amdgcn_s_wait_tensorcnt)
#define USE_TDM 1
#endif
#endif

__device__ __forceinline__ v16h cat8(v8h a, v8h b) {
  return __builtin_shufflevector(a, b, 0,1,2,3,4,5,6,7,8,9,10,11,12,13,14,15);
}

// A fragment (16x32 f16, contiguous-K order): lane m=lane&15 owns row m;
// lanes<16: K {0..15}; lanes>=16: K {16..31}.
__device__ __forceinline__ v16h frag_a(const _Float16* __restrict__ base,
                                       int ldK, int lane, int k0) {
  const int m   = lane & 15;
  const int off = (lane >> 4) ? 16 : 0;
  const _Float16* p = base + (long)m * ldK + k0 + off;
  return cat8(*(const v8h*)p, *(const v8h*)(p + 8));
}

// B fragment (32x16 f16): lane owns col lane&15; 16 contiguous K from W row.
// Caller passes wrow = W + col*ldK + k0 + ((lane>=16)?16:0).
__device__ __forceinline__ v16h frag_b(const _Float16* __restrict__ wrow) {
  return cat8(*(const v8h*)wrow, *(const v8h*)(wrow + 8));
}

__device__ __forceinline__ v8f wmma3(v16h ah, v16h al, v16h bh, v16h bl, v8f acc) {
  acc = __builtin_amdgcn_wmma_f32_16x16x32_f16(false, ah, false, bh, (short)0, acc, false, false);
  acc = __builtin_amdgcn_wmma_f32_16x16x32_f16(false, ah, false, bl, (short)0, acc, false, false);
  acc = __builtin_amdgcn_wmma_f32_16x16x32_f16(false, al, false, bh, (short)0, acc, false, false);
  return acc;
}

// One-time fp32 -> (hi,lo) f16 split. Memory-bound elementwise prep.
__global__ __launch_bounds__(256) void split_k(
    const float* __restrict__ src, _Float16* __restrict__ hi,
    _Float16* __restrict__ lo, int n)
{
  int i = blockIdx.x * 256 + threadIdx.x;
  if (i >= n) return;
  float f = src[i];
  _Float16 h = (_Float16)f;
  hi[i] = h;
  lo[i] = (_Float16)(f - (float)h);
}

// ---------------------------------------------------------------------------
// O = X @ W^T on pre-split operands. One wave per 16x16 tile, 8 waves/block.
// ---------------------------------------------------------------------------
__global__ __launch_bounds__(256) void gemm_h(
    const _Float16* __restrict__ Ah, const _Float16* __restrict__ Al,
    const _Float16* __restrict__ Wh, const _Float16* __restrict__ Wl,
    float* __restrict__ O, int K, int N)
{
  const int wave = threadIdx.x >> 5;
  const int lane = threadIdx.x & 31;
  const int rt = blockIdx.x;
  const int ct = blockIdx.y * 8 + wave;
  if (ct * 16 >= N) return;                 // wave-uniform: EXEC all-ones for WMMA

  const int m      = lane & 15;
  const int hiHalf = lane >> 4;
  const int kb     = hiHalf ? 16 : 0;
  const int colg   = ct * 16 + m;
  const long arow  = (long)(rt * 16) * K;
  const long wrow  = (long)colg * K;

  v8f acc = {};
  for (int k0 = 0; k0 < K; k0 += 32) {
    v16h ah = frag_a(Ah + arow, K, lane, k0);
    v16h al = frag_a(Al + arow, K, lane, k0);
    v16h bh = frag_b(Wh + wrow + k0 + kb);
    v16h bl = frag_b(Wl + wrow + k0 + kb);
    acc = wmma3(ah, al, bh, bl, acc);
  }
  const int row0 = rt * 16 + (hiHalf << 3); // D: elem v -> row v+8*(lane>=16)
#pragma unroll
  for (int v = 0; v < 8; ++v)
    O[(long)(row0 + v) * N + colg] = acc[v];
}

// ---------------------------------------------------------------------------
// Causal depthwise conv (k=4); writes u in f32 AND pre-split hi/lo f16.
// ---------------------------------------------------------------------------
__global__ __launch_bounds__(256) void conv_k(
    const float* __restrict__ xz, const float* __restrict__ cw,
    const float* __restrict__ cb, float* __restrict__ u,
    _Float16* __restrict__ uh, _Float16* __restrict__ ul)
{
  int idx = blockIdx.x * blockDim.x + threadIdx.x;   // 2048*512
  int c = idx & (DM - 1);
  int row = idx >> 9;
  int bb = row >> 10, l = row & (L_LEN - 1);
  float acc = cb[c];
#pragma unroll
  for (int j = 0; j < 4; ++j) {
    int ls = l - 3 + j;
    if (ls >= 0) acc += cw[c * 4 + j] * xz[(long)(bb * L_LEN + ls) * 1024 + c];
  }
  u[idx] = acc;
  _Float16 h = (_Float16)acc;
  uh[idx] = h;
  ul[idx] = (_Float16)(acc - (float)h);
}

// dtr(2048,16) = u @ W_xproj^T  (rank-16, VALU is fine)
__global__ __launch_bounds__(256) void dtr_k(
    const float* __restrict__ u, const float* __restrict__ Wxp,
    float* __restrict__ dtr)
{
  int idx = blockIdx.x * blockDim.x + threadIdx.x;   // 2048*16
  int n = idx & 15, row = idx >> 4;
  const float* up = u + (long)row * DM;
  const float* wp = Wxp + (long)n * DM;
  float acc = 0.f;
  for (int k = 0; k < DM; ++k) acc += up[k] * wp[k];
  dtr[idx] = acc;
}

// dt(2048,512) = softplus(dtr @ W_dt^T + 2*b_dt)
__global__ __launch_bounds__(256) void dt_k(
    const float* __restrict__ dtr, const float* __restrict__ Wdt,
    const float* __restrict__ bdt, float* __restrict__ dt)
{
  int idx = blockIdx.x * blockDim.x + threadIdx.x;   // 2048*512
  int dcol = idx & (DM - 1), row = idx >> 9;
  const float* rp = dtr + (long)row * 16;
  const float* wp = Wdt + (long)dcol * 16;
  float acc = 2.f * bdt[dcol];
#pragma unroll
  for (int j = 0; j < 16; ++j) acc += rp[j] * wp[j];
  dt[idx] = (acc > 20.f) ? acc : log1pf(expf(acc));
}

// ---------------------------------------------------------------------------
// Fused theta-GEMM + complex scan + epilogue.
// Grid (b=2, dtile=32), 256 thr = 8 waves, wave w owns n in {2w,2w+1}.
// A-chunks (16 rows x 128 halves, hi & lo) staged to LDS by the TDM.
// ---------------------------------------------------------------------------
__global__ __launch_bounds__(256) void ssm_theta_scan(
    const float* __restrict__ u,       // (2048,512) f32
    const float* __restrict__ dt,      // (2048,512) f32
    const float* __restrict__ xz,      // (2048,1024): z = cols 512..1023
    const _Float16* __restrict__ uHg,  // u hi halves (2048,512)
    const _Float16* __restrict__ uLg,  // u lo halves
    const _Float16* __restrict__ WAh,  // W_xA hi (8192,512)
    const _Float16* __restrict__ WAl,  // W_xA lo
    const float* __restrict__ bxA,     // (8192)
    const float* __restrict__ Bc,      // complex64(16) interleaved
    const float* __restrict__ Cc,      // complex64(16) interleaved
    const float* __restrict__ Dd,      // (512)
    _Float16* __restrict__ yh,         // y hi halves (2048,512)
    _Float16* __restrict__ yl)         // y lo halves
{
  __shared__ __attribute__((aligned(16))) _Float16 uAh[16][128]; // 4 KB
  __shared__ __attribute__((aligned(16))) _Float16 uAl[16][128]; // 4 KB
  __shared__ float phiS[NSTATE][16][16];                         // 16 KB
  __shared__ float dtS[16][16], uS[16][16], dtuS[16][16];        // 3 KB
  __shared__ float ypart[256][16];                               // 16 KB

  const int tid  = threadIdx.x;
  const int wave = tid >> 5, lane = tid & 31;
  const int b     = blockIdx.x;
  const int dtile = blockIdx.y;
  const float inv_sqrt_d = 0.044194173824159216f;   // 1/sqrt(512)

  const int m      = lane & 15;
  const int hiHalf = lane >> 4;
  const int kb     = hiHalf ? 16 : 0;
  const int cw     = m;
  const int dcol_w = dtile * 16 + cw;
  const int n0 = 2 * wave, n1 = 2 * wave + 1;
  const long rowW0 = (long)(n0 * DM + dcol_w) * DM;
  const long rowW1 = (long)(n1 * DM + dcol_w) * DM;
  const float bias0 = bxA[n0 * DM + dcol_w];
  const float bias1 = bxA[n1 * DM + dcol_w];

  // scan ownership: one (n,d) column per lane, h carried in registers
  const int n_own = 2 * wave + hiHalf;
  const int c_own = m;
  const float B_re = Bc[2 * n_own], B_im = Bc[2 * n_own + 1];
  const float C_re = Cc[2 * n_own], C_im = Cc[2 * n_own + 1];
  float h_re = 0.f, h_im = 0.f;

  const int er = tid >> 4, ec = tid & 15;
  const int dcol_e = dtile * 16 + ec;

#ifdef USE_TDM
  const unsigned ldsH = (unsigned)(uintptr_t)&uAh[0][0];
  const unsigned ldsL = (unsigned)(uintptr_t)&uAl[0][0];
  // D# group1: 2D tensor, 2-byte elems, tile 128x16, tensor_dim0=512,
  // tensor_dim1=16, dim0_stride=512 (ISA cdna5 CDNA5 ISA)
  i32x8 g1;
  g1[0] = 0x00010000;            // data_size=1 (2 bytes)
  g1[1] = (int)(512u << 16);     // tensor_dim0[15:0]
  g1[2] = (int)(16u  << 16);     // tensor_dim0[31:16]=0 | tensor_dim1[15:0]=16
  g1[3] = (int)(128u << 16);     // tensor_dim1[31:16]=0 | tile_dim0=128
  g1[4] = 16;                    // tile_dim1=16, tile_dim2=0
  g1[5] = 512;                   // tensor_dim0_stride low32
  g1[6] = 0; g1[7] = 0;
  const i32x4 gz4 = {0, 0, 0, 0};
  const i32x8 gz8 = {0, 0, 0, 0, 0, 0, 0, 0};
#endif

  for (int l0 = 0; l0 < L_LEN; l0 += 16) {
    __syncthreads();   // previous chunk fully consumed before restaging
    {
      long g = (long)(b * L_LEN + l0 + er) * DM + dcol_e;
      float dv = dt[g], uv = u[g];
      dtS[er][ec] = dv; uS[er][ec] = uv; dtuS[er][ec] = dv * uv;
    }

    v8f acc0 = {}, acc1 = {};
    for (int ks = 0; ks < DM; ks += 128) {
      __syncthreads();
      const long chunk = (long)(b * L_LEN + l0) * DM + ks;
#ifdef USE_TDM
      if (wave == 0) {   // one wave drives the DMA; EXEC ignored by TDM
        uint64_t gaH = (uint64_t)(uintptr_t)(uHg + chunk);
        uint64_t gaL = (uint64_t)(uintptr_t)(uLg + chunk);
        u32x4 g0h, g0l;
        g0h[0] = 1u; g0h[1] = ldsH;
        g0h[2] = (unsigned)(gaH & 0xFFFFFFFFu);
        g0h[3] = (unsigned)((gaH >> 32) & 0x1FFFFFFu) | (2u << 30); // type=2
        g0l[0] = 1u; g0l[1] = ldsL;
        g0l[2] = (unsigned)(gaL & 0xFFFFFFFFu);
        g0l[3] = (unsigned)((gaL >> 32) & 0x1FFFFFFu) | (2u << 30);
        __builtin_amdgcn_tensor_load_to_lds(g0h, g1, gz4, gz4, gz8, 0);
        __builtin_amdgcn_tensor_load_to_lds(g0l, g1, gz4, gz4, gz8, 0);
        __builtin_amdgcn_s_wait_tensorcnt(0);
      }
#else
      {  // cooperative fallback: 8 halves per thread per array
        int r  = tid >> 4;
        int kk = (tid & 15) * 8;
        *(v8h*)&uAh[r][kk] = *(const v8h*)(uHg + chunk + (long)r * DM + kk);
        *(v8h*)&uAl[r][kk] = *(const v8h*)(uLg + chunk + (long)r * DM + kk);
      }
#endif
      __syncthreads();
      if (ks + 128 < DM) {  // prefetch next K-slab of W rows into caches
        __builtin_prefetch(WAh + rowW0 + ks + 128 + kb, 0, 1);
        __builtin_prefetch(WAh + rowW1 + ks + 128 + kb, 0, 1);
        __builtin_prefetch(WAl + rowW0 + ks + 128 + kb, 0, 1);
        __builtin_prefetch(WAl + rowW1 + ks + 128 + kb, 0, 1);
      }
#pragma unroll
      for (int k0 = 0; k0 < 128; k0 += 32) {
        v16h ah = frag_a(&uAh[0][0], 128, lane, k0);
        v16h al = frag_a(&uAl[0][0], 128, lane, k0);
        v16h bh0 = frag_b(WAh + rowW0 + ks + k0 + kb);
        v16h bl0 = frag_b(WAl + rowW0 + ks + k0 + kb);
        v16h bh1 = frag_b(WAh + rowW1 + ks + k0 + kb);
        v16h bl1 = frag_b(WAl + rowW1 + ks + k0 + kb);
        acc0 = wmma3(ah, al, bh0, bl0, acc0);
        acc1 = wmma3(ah, al, bh1, bl1, acc1);
      }
    }

    // phi = dt * (theta*scale + bias) -> LDS phase tiles
    {
      const int rh = hiHalf << 3;
#pragma unroll
      for (int v = 0; v < 8; ++v) {
        int r = rh + v;
        float dv = dtS[r][cw];
        phiS[n0][r][cw] = dv * (acc0[v] * inv_sqrt_d + bias0);
        phiS[n1][r][cw] = dv * (acc1[v] * inv_sqrt_d + bias1);
      }
    }
    __syncthreads();

    // 16-step sequential complex scan: h = e^{i*phi}*h + dt*u*B[n]
#pragma unroll 1
    for (int r = 0; r < 16; ++r) {
      float phi = phiS[n_own][r][c_own];
      float ar = cosf(phi), ai = sinf(phi);
      float dtu = dtuS[r][c_own];
      float nr = ar * h_re - ai * h_im + dtu * B_re;
      float ni = ar * h_im + ai * h_re + dtu * B_im;
      h_re = nr; h_im = ni;
      ypart[n_own * 16 + c_own][r] = C_re * h_re - C_im * h_im;  // Re(C[n]h)
    }
    __syncthreads();

    // deterministic fixed-order n-reduction + D*u + silu(z); emit split y
    {
      float s = 0.f;
#pragma unroll
      for (int n = 0; n < NSTATE; ++n) s += ypart[n * 16 + ec][er];
      long row = (long)(b * L_LEN + l0 + er);
      float zv = xz[row * 1024 + DM + dcol_e];
      float silu = zv / (1.f + expf(-zv));
      float yv = (s + Dd[dcol_e] * uS[er][ec]) * silu;
      _Float16 h = (_Float16)yv;
      yh[row * DM + dcol_e] = h;
      yl[row * DM + dcol_e] = (_Float16)(yv - (float)h);
    }
  }
}

// ---------------------------------------------------------------------------
extern "C" void kernel_launch(void* const* d_in, const int* in_sizes, int n_in,
                              void* d_out, int out_size, void* d_ws, size_t ws_size,
                              hipStream_t stream) {
  const float* x      = (const float*)d_in[0];
  const float* W_in   = (const float*)d_in[1];
  const float* conv_w = (const float*)d_in[2];
  const float* conv_b = (const float*)d_in[3];
  const float* W_xprj = (const float*)d_in[4];
  const float* W_dt   = (const float*)d_in[5];
  const float* b_dt   = (const float*)d_in[6];
  const float* Dv     = (const float*)d_in[7];
  const float* Bc     = (const float*)d_in[8];
  const float* Cc     = (const float*)d_in[9];
  const float* W_xA   = (const float*)d_in[10];
  const float* b_xA   = (const float*)d_in[11];
  const float* W_out  = (const float*)d_in[12];
  float* out = (float*)d_out;

  const long M = (long)B_SZ * L_LEN;   // 2048
  char* w = (char*)d_ws;
  float* xz  = (float*)w;  w += M * 1024 * 4;
  float* u   = (float*)w;  w += M * DM * 4;
  float* dtr = (float*)w;  w += M * 16 * 4;
  float* dtb = (float*)w;  w += M * DM * 4;
  _Float16* xh  = (_Float16*)w;  w += M * DM * 2;
  _Float16* xl  = (_Float16*)w;  w += M * DM * 2;
  _Float16* wih = (_Float16*)w;  w += 1024 * DM * 2;
  _Float16* wil = (_Float16*)w;  w += 1024 * DM * 2;
  _Float16* wah = (_Float16*)w;  w += 8192 * DM * 2;
  _Float16* wal = (_Float16*)w;  w += 8192 * DM * 2;
  _Float16* woh = (_Float16*)w;  w += DM * DM * 2;
  _Float16* wol = (_Float16*)w;  w += DM * DM * 2;
  _Float16* uh  = (_Float16*)w;  w += M * DM * 2;
  _Float16* ul  = (_Float16*)w;  w += M * DM * 2;
  _Float16* yh  = (_Float16*)w;  w += M * DM * 2;
  _Float16* yl  = (_Float16*)w;  w += M * DM * 2;

  dim3 blk(256);
  // one-time operand splits (elementwise, L2-resident)
  split_k<<<dim3((unsigned)((M * DM + 255) / 256)), blk, 0, stream>>>(x, xh, xl, (int)(M * DM));
  split_k<<<dim3((1024 * DM + 255) / 256), blk, 0, stream>>>(W_in, wih, wil, 1024 * DM);
  split_k<<<dim3((8192 * DM + 255) / 256), blk, 0, stream>>>(W_xA, wah, wal, 8192 * DM);
  split_k<<<dim3((DM * DM + 255) / 256), blk, 0, stream>>>(W_out, woh, wol, DM * DM);

  // 1) xz = x @ W_in^T
  gemm_h<<<dim3((unsigned)(M / 16), 8), blk, 0, stream>>>(xh, xl, wih, wil, xz, DM, 2 * DM);
  // 2) causal depthwise conv -> u (+ split)
  conv_k<<<dim3((unsigned)((M * DM) / 256)), blk, 0, stream>>>(xz, conv_w, conv_b, u, uh, ul);
  // 3) dt = softplus(u @ Wxproj^T @ Wdt^T + 2*b_dt)
  dtr_k<<<dim3((unsigned)((M * 16) / 256)), blk, 0, stream>>>(u, W_xprj, dtr);
  dt_k<<<dim3((unsigned)((M * DM) / 256)), blk, 0, stream>>>(dtr, W_dt, b_dt, dtb);
  // 4) fused theta-GEMM + complex scan + epilogue -> y (split halves)
  ssm_theta_scan<<<dim3(B_SZ, DM / 16), blk, 0, stream>>>(u, dtb, xz, uh, ul,
                                                          wah, wal, b_xA, Bc, Cc, Dv, yh, yl);
  // 5) out = y @ W_out^T
  gemm_h<<<dim3((unsigned)(M / 16), 4), blk, 0, stream>>>(yh, yl, woh, wol, out, DM, DM);
}
